// Attention_5987184410876
// MI455X (gfx1250) — compile-verified, matrix operand order checked by reading, not run on hardware
//
#include <hip/hip_runtime.h>

// ---------------------------------------------------------------------------
// Problem constants (from reference)
// ---------------------------------------------------------------------------
constexpr int ENC_DIM = 2048;
constexpr int ATT_DIM = 512;
constexpr int BB      = 256;
constexpr int PP      = 196;
constexpr int ROWS    = BB * PP;   // 50176 flat rows of encoder_out
constexpr int MT      = 32;        // rows per workgroup in the main GEMM
constexpr int KC      = 32;        // K chunk (bf16 WMMA K)
constexpr int LDA     = 40;        // padded LDS row (elements) -> 80B, 16B aligned
constexpr int LDB     = 40;

typedef __attribute__((ext_vector_type(8)))  float  v8f;
typedef __attribute__((ext_vector_type(16))) __bf16 v16bf;
typedef __attribute__((ext_vector_type(8)))  __bf16 v8bf;
typedef __attribute__((ext_vector_type(4)))  __bf16 bf16x4;
typedef __attribute__((ext_vector_type(2)))  __bf16 bf16x2;

union FragBf {
    v16bf v;
    v8bf  h[2];
};

// ---------------------------------------------------------------------------
// Kernel 1: att2[b,a] = dh[b,:] @ Wd[:,a] + Wd_b[a] + We_b[a]
// ---------------------------------------------------------------------------
__global__ __launch_bounds__(256)
void att2_kernel(const float* __restrict__ dh, const float* __restrict__ Wd,
                 const float* __restrict__ Wd_b, const float* __restrict__ We_b,
                 float* __restrict__ att2) {
    __shared__ float h[512];
    const int b = blockIdx.x;
    const int t = threadIdx.x;
    h[t]       = dh[b * 512 + t];
    h[t + 256] = dh[b * 512 + t + 256];
    __syncthreads();
    #pragma unroll
    for (int half = 0; half < 2; ++half) {
        const int a = t + half * 256;
        float acc = Wd_b[a] + We_b[a];
        for (int d = 0; d < 512; ++d)
            acc = fmaf(h[d], Wd[d * 512 + a], acc);   // coalesced across lanes
        att2[b * 512 + a] = acc;
    }
}

// ---------------------------------------------------------------------------
// Kernel 2 (main): bf16 WMMA GEMM  (50176 x 2048) @ (2048 x 512)
// fused with relu(att1 + att2) . Wf  ->  score[row]
// 8 waves: mw in {0,1} (16-row tiles), nw in {0..3} (128-col groups of 8 tiles)
// ---------------------------------------------------------------------------
__global__ __launch_bounds__(256)
void att_score_kernel(const float* __restrict__ enc,
                      const float* __restrict__ We,
                      const float* __restrict__ att2,
                      const float* __restrict__ Wf,
                      float* __restrict__ score_out) {
    __shared__ __align__(16) __bf16 Alds[MT * LDA];        //  2.5 KB (32 rows x 32 k)
    __shared__ __align__(16) __bf16 Blds[ATT_DIM * LDB];   // 40.0 KB (n-major: [n][k])
    __shared__ float score[MT];

    const int tid  = threadIdx.x;
    const int lane = tid & 31;
    const int wave = tid >> 5;
    const int mw   = wave >> 2;   // 0..1
    const int nw   = wave & 3;    // 0..3
    const int row0 = blockIdx.x * MT;
    const int l15  = lane & 15;
    const int hi   = lane >> 4;   // 0 or 1 (lane half)
    const int kb   = hi << 3;     // K offset 0 or 8 per fragment layout

    if (tid < MT) score[tid] = 0.0f;

    v8f acc[8];
    #pragma unroll
    for (int t = 0; t < 8; ++t) acc[t] = v8f{};

    // A-staging coords (encoder rows: the HBM stream)
    const int arow = tid >> 3;            // 0..31
    const int kq   = (tid & 7) << 2;      // 0..28 step 4
    const float* aptr = enc + (size_t)(row0 + arow) * ENC_DIM + kq;

    for (int k0 = 0; k0 < ENC_DIM; k0 += KC) {
        // ---- stage A: 32 rows x 32 k, fp32 -> bf16 (native cvt), float4 reads ----
        {
            const float4 f = *(const float4*)(aptr + k0);
            bf16x4 c = { (__bf16)f.x, (__bf16)f.y, (__bf16)f.z, (__bf16)f.w };
            *(bf16x4*)&Alds[arow * LDA + kq] = c;
        }
        // prefetch next A chunk while WMMA phase runs (global_prefetch_b8)
        if (k0 + KC < ENC_DIM)
            __builtin_prefetch(aptr + k0 + KC, 0, 3);

        // ---- stage B: 32 k x 512 n, k-pair reads, transposed b32 writes [n][k] ----
        #pragma unroll
        for (int it = 0; it < 8; ++it) {
            const int f  = it * 256 + tid;     // 0..2047 (float4 x k-pair units)
            const int k  = (f >> 7) << 1;      // even k: 0..30
            const int n  = (f & 127) << 2;     // 0..508
            const float4 w0 = *(const float4*)(We + (size_t)(k0 + k)     * ATT_DIM + n);
            const float4 w1 = *(const float4*)(We + (size_t)(k0 + k + 1) * ATT_DIM + n);
            bf16x2 p0 = { (__bf16)w0.x, (__bf16)w1.x };
            bf16x2 p1 = { (__bf16)w0.y, (__bf16)w1.y };
            bf16x2 p2 = { (__bf16)w0.z, (__bf16)w1.z };
            bf16x2 p3 = { (__bf16)w0.w, (__bf16)w1.w };
            *(bf16x2*)&Blds[(n + 0) * LDB + k] = p0;
            *(bf16x2*)&Blds[(n + 1) * LDB + k] = p1;
            *(bf16x2*)&Blds[(n + 2) * LDB + k] = p2;
            *(bf16x2*)&Blds[(n + 3) * LDB + k] = p3;
        }
        __syncthreads();

        // ---- A fragment: lane = row, two contiguous 16B halves (K and K+16) ----
        FragBf a;
        {
            const int ar = mw * 16 + l15;
            a.h[0] = *(const v8bf*)&Alds[ar * LDA + kb];
            a.h[1] = *(const v8bf*)&Alds[ar * LDA + kb + 16];
        }
        // ---- 8 N-tiles, B fragments ping-ponged so ds_load overlaps WMMA ----
        FragBf bcur, bnext;
        {
            const int n = nw * 128 + l15;
            bcur.h[0] = *(const v8bf*)&Blds[n * LDB + kb];
            bcur.h[1] = *(const v8bf*)&Blds[n * LDB + kb + 16];
        }
        #pragma unroll
        for (int t = 0; t < 8; ++t) {
            if (t < 7) {
                const int n = nw * 128 + (t + 1) * 16 + l15;
                bnext.h[0] = *(const v8bf*)&Blds[n * LDB + kb];
                bnext.h[1] = *(const v8bf*)&Blds[n * LDB + kb + 16];
            }
            acc[t] = __builtin_amdgcn_wmma_f32_16x16x32_bf16(
                false, a.v, false, bcur.v, (short)0, acc[t], false, false);
            bcur = bnext;
        }
        __syncthreads();
    }

    // ---- epilogue: relu(c + att2[b,n]) * Wf[n], reduce over this wave's 128 cols ----
    float partial[8];
    #pragma unroll
    for (int v = 0; v < 8; ++v) partial[v] = 0.0f;

    #pragma unroll
    for (int t = 0; t < 8; ++t) {
        const int n = nw * 128 + t * 16 + l15;
        const float wfv = Wf[n];
        #pragma unroll
        for (int v = 0; v < 8; ++v) {
            const int mrow = v + (hi << 3);            // C layout: VGPR v -> row v / v+8
            const int r = row0 + mw * 16 + mrow;
            const int b = r / PP;
            float val = acc[t][v] + att2[b * ATT_DIM + n];
            val = fmaxf(val, 0.0f);
            partial[v] = fmaf(val, wfv, partial[v]);
        }
    }
    // reduce across the 16 lanes holding one row (xor 1,2,4,8 stays within halves)
    #pragma unroll
    for (int v = 0; v < 8; ++v) {
        float s = partial[v];
        s += __shfl_xor(s, 1, 32);
        s += __shfl_xor(s, 2, 32);
        s += __shfl_xor(s, 4, 32);
        s += __shfl_xor(s, 8, 32);
        if (l15 == 0) {
            const int mrow = v + (hi << 3);
            atomicAdd(&score[mw * 16 + mrow], s);      // ds_add_f32, 4 nw-groups combine
        }
    }
    __syncthreads();
    if (tid < MT) score_out[row0 + tid] = score[tid];  // flat r = b*196+p
}

// ---------------------------------------------------------------------------
// Kernel 3: in-place softmax over P=196 per batch row (Wf_b dropped: shift-invariant)
// ---------------------------------------------------------------------------
__global__ __launch_bounds__(256)
void softmax_kernel(float* __restrict__ alpha) {
    __shared__ float red[256];
    const int b = blockIdx.x;
    const int t = threadIdx.x;
    float* row = alpha + b * PP;
    const float x = (t < PP) ? row[t] : -3.402823466e38f;
    red[t] = x;
    __syncthreads();
    for (int s = 128; s > 0; s >>= 1) {
        if (t < s) red[t] = fmaxf(red[t], red[t + s]);
        __syncthreads();
    }
    const float mx = red[0];
    __syncthreads();
    const float e = (t < PP) ? expf(x - mx) : 0.0f;
    red[t] = e;
    __syncthreads();
    for (int s = 128; s > 0; s >>= 1) {
        if (t < s) red[t] += red[t + s];
        __syncthreads();
    }
    const float inv = 1.0f / red[0];
    if (t < PP) row[t] = e * inv;
}

// ---------------------------------------------------------------------------
// Kernel 4: awe[b,e] = sum_p enc[b,p,e] * alpha[b,p]  (bandwidth-bound GEMV)
// 512 blocks: (b, e-chunk of 1024); thread owns 4 e's (float4)
// ---------------------------------------------------------------------------
__global__ __launch_bounds__(256)
void awe_kernel(const float* __restrict__ enc, const float* __restrict__ alpha,
                float* __restrict__ awe) {
    __shared__ float al[PP];
    const int b     = blockIdx.x >> 1;
    const int chunk = blockIdx.x & 1;
    const int t     = threadIdx.x;
    if (t < PP) al[t] = alpha[b * PP + t];
    __syncthreads();
    const int e = chunk * 1024 + t * 4;
    float4 acc = make_float4(0.f, 0.f, 0.f, 0.f);
    const float* base = enc + (size_t)b * PP * ENC_DIM + e;
    for (int p = 0; p < PP; ++p) {
        const float4 v = *(const float4*)(base + (size_t)p * ENC_DIM);
        const float a = al[p];
        acc.x = fmaf(v.x, a, acc.x);
        acc.y = fmaf(v.y, a, acc.y);
        acc.z = fmaf(v.z, a, acc.z);
        acc.w = fmaf(v.w, a, acc.w);
    }
    *(float4*)(awe + (size_t)b * ENC_DIM + e) = acc;
}

// ---------------------------------------------------------------------------
// Launch
// ---------------------------------------------------------------------------
extern "C" void kernel_launch(void* const* d_in, const int* in_sizes, int n_in,
                              void* d_out, int out_size, void* d_ws, size_t ws_size,
                              hipStream_t stream) {
    const float* enc  = (const float*)d_in[0];   // (256,196,2048)
    const float* dh   = (const float*)d_in[1];   // (1,256,512)
    const float* We_w = (const float*)d_in[2];   // (2048,512)
    const float* We_b = (const float*)d_in[3];   // (512,)
    const float* Wd_w = (const float*)d_in[4];   // (512,512)
    const float* Wd_b = (const float*)d_in[5];   // (512,)
    const float* Wf_w = (const float*)d_in[6];   // (512,)
    // d_in[7] = Wf_b: added after the relu-dot -> softmax shift-invariant -> unused

    float* out   = (float*)d_out;
    float* awe   = out;                 // 256*2048 floats (first output)
    float* alpha = out + BB * ENC_DIM;  // 256*196  floats (second output)
    float* att2  = (float*)d_ws;        // 256*512 floats of scratch

    att2_kernel<<<BB, 256, 0, stream>>>(dh, Wd_w, Wd_b, We_b, att2);
    att_score_kernel<<<ROWS / MT, 256, 0, stream>>>(enc, We_w, att2, Wf_w, alpha);
    softmax_kernel<<<BB, 256, 0, stream>>>(alpha);
    awe_kernel<<<BB * 2, 256, 0, stream>>>(enc, alpha, awe);
}
